// SEE_5102421147974
// MI455X (gfx1250) — compile-verified, matrix operand order checked by reading, not run on hardware
//
#include <hip/hip_runtime.h>
#include <hip/hip_bf16.h>
#include <cstdint>

typedef __attribute__((ext_vector_type(16))) _Float16 v16h;
typedef __attribute__((ext_vector_type(8)))  _Float16 v8h;
typedef __attribute__((ext_vector_type(8)))  float    v8f;

#define B_  64
#define T_  1000
#define F_  140
#define D_  256
#define KPAD1 160          // F padded to multiple of 32
#define M_  (B_ * T_)      // 64000 rows, index m = t*64 + b
#define NBD (B_ * D_)      // 16384
#define TAU_INV 0.5f
#define V_TH 1.0f
#define EPS_ 1e-5f

// ---------- fragment helpers (16-bit A/B layout, wave32) ----------
// lane L holds row (L&15); K elements: [0..7] = kc + (L>>4)*8 .. +7,
//                            [8..15] = kc + 16 + (L>>4)*8 .. +7
__device__ inline v16h load_frag(const _Float16* __restrict__ p) {
    v8h lo = *(const v8h*)(p);
    v8h hi = *(const v8h*)(p + 16);
    v16h r;
#pragma unroll
    for (int i = 0; i < 8; ++i) { r[i] = lo[i]; r[i + 8] = hi[i]; }
    return r;
}
__device__ inline v8f zero8() {
    v8f z;
#pragma unroll
    for (int i = 0; i < 8; ++i) z[i] = 0.0f;
    return z;
}

// ---------- conversion kernels ----------
// xh[m = t*64+b][f(0..159)] = f16(x[b][t][f]), zero-padded f>=140
__global__ void convert_x_kernel(const float* __restrict__ x, _Float16* __restrict__ xh) {
    int tid = blockIdx.x * blockDim.x + threadIdx.x;
    const int n = M_ * KPAD1;
    if (tid >= n) return;
    int m = tid / KPAD1;
    int f = tid - m * KPAD1;
    int t = m >> 6;          // m / 64
    int b = m & 63;
    float v = 0.0f;
    if (f < F_) v = x[(size_t)b * (T_ * F_) + (size_t)t * F_ + f];
    xh[tid] = (_Float16)v;
}

// pw_h[d][f(0..159)] padded; lin_h[e][d] straight convert
__global__ void convert_w_kernel(const float* __restrict__ pw_w, const float* __restrict__ lin_w,
                                 _Float16* __restrict__ pwh, _Float16* __restrict__ linh) {
    int tid = blockIdx.x * blockDim.x + threadIdx.x;
    const int npw = D_ * KPAD1;       // 40960
    const int nlin = D_ * D_;         // 65536
    if (tid < npw) {
        int d = tid / KPAD1;
        int f = tid - d * KPAD1;
        float v = (f < F_) ? pw_w[d * F_ + f] : 0.0f;
        pwh[tid] = (_Float16)v;
    }
    if (tid < nlin) {
        linh[tid] = (_Float16)lin_w[tid];
    }
}

// ---------- GEMM1: h[m][d] = f16(sum_f xh[m][f]*pwh[d][f] + pw_b[d]) ----------
// one wave -> 32x64 tile (2 row-blocks share each B fragment)
// waves = (M/32) * (256/64) = 8000 -> 1000 blocks of 8 waves
__global__ void gemm1_kernel(const _Float16* __restrict__ xh, const _Float16* __restrict__ pwh,
                             const float* __restrict__ pwb, _Float16* __restrict__ h) {
    const int lane = threadIdx.x & 31;
    const int wave = blockIdx.x * (blockDim.x >> 5) + (threadIdx.x >> 5);
    const int ng = wave & 3;
    const int mt = wave >> 2;
    const int m0 = mt * 32;
    const int n0 = ng * 64;
    const int half = lane >> 4;
    const int l15 = lane & 15;

    v8f acc0[4], acc1[4];
#pragma unroll
    for (int i = 0; i < 4; ++i) { acc0[i] = zero8(); acc1[i] = zero8(); }

    const _Float16* a0ptr = xh + (size_t)(m0 + l15) * KPAD1 + half * 8;
    const _Float16* a1ptr = a0ptr + (size_t)16 * KPAD1;
#pragma unroll
    for (int kc = 0; kc < KPAD1; kc += 32) {
        v16h a0 = load_frag(a0ptr + kc);
        v16h a1 = load_frag(a1ptr + kc);
#pragma unroll
        for (int nt = 0; nt < 4; ++nt) {
            const _Float16* bptr = pwh + (size_t)(n0 + nt * 16 + l15) * KPAD1 + half * 8 + kc;
            v16h bf = load_frag(bptr);
            acc0[nt] = __builtin_amdgcn_wmma_f32_16x16x32_f16(
                false, a0, false, bf, (short)0, acc0[nt], false, false);
            acc1[nt] = __builtin_amdgcn_wmma_f32_16x16x32_f16(
                false, a1, false, bf, (short)0, acc1[nt], false, false);
        }
    }
#pragma unroll
    for (int nt = 0; nt < 4; ++nt) {
        int col = n0 + nt * 16 + l15;
        float bbias = pwb[col];
#pragma unroll
        for (int r = 0; r < 8; ++r) {
            int row0 = m0 + r + 8 * half;
            h[(size_t)row0 * D_ + col]        = (_Float16)(acc0[nt][r] + bbias);
            h[(size_t)(row0 + 16) * D_ + col] = (_Float16)(acc1[nt][r] + bbias);
        }
    }
}

// ---------- depthwise conv (K=7, same pad) + BN1 + LIF1 scan ----------
// thread tid = b*256 + d; sequential over t in groups of 4; h layout (T, B*D) f16
__global__ void dw_lif1_kernel(const _Float16* __restrict__ h, const float* __restrict__ dww,
                               const float* __restrict__ g, const float* __restrict__ bb,
                               const float* __restrict__ mm, const float* __restrict__ vv,
                               _Float16* __restrict__ s1) {
    int tid = blockIdx.x * blockDim.x + threadIdx.x;
    if (tid >= NBD) return;
    int d = tid & (D_ - 1);

    float wk[7];
#pragma unroll
    for (int k = 0; k < 7; ++k) wk[k] = dww[d * 7 + k];
    float scale = g[d] * rsqrtf(vv[d] + EPS_);
    float bias  = bb[d] - mm[d] * scale;

    float win[7];
#pragma unroll
    for (int k = 0; k < 7; ++k) win[k] = 0.0f;
    win[3] = (float)h[(size_t)0 * NBD + tid];
    win[4] = (float)h[(size_t)1 * NBD + tid];
    win[5] = (float)h[(size_t)2 * NBD + tid];

    float v = 0.0f;
    for (int t0 = 0; t0 < T_; t0 += 4) {
        // batch 4 independent loads (one clause) + prefetch next group
        float hn[4];
#pragma unroll
        for (int j = 0; j < 4; ++j) {
            int tl = t0 + j + 3;
            hn[j] = (tl < T_) ? (float)h[(size_t)tl * NBD + tid] : 0.0f;
        }
        if (t0 + 11 < T_)
            __builtin_prefetch(h + (size_t)(t0 + 11) * NBD + tid, 0, 1);
#pragma unroll
        for (int j = 0; j < 4; ++j) {
            win[6] = hn[j];
            float conv = 0.0f;
#pragma unroll
            for (int k = 0; k < 7; ++k) conv = fmaf(win[k], wk[k], conv);
            float xx = conv * scale + bias;
            v = v + (xx - v) * TAU_INV;
            bool sp = (v - V_TH) >= 0.0f;
            s1[(size_t)(t0 + j) * NBD + tid] = (_Float16)(sp ? 1.0f : 0.0f);
            v = sp ? 0.0f : v;
#pragma unroll
            for (int k = 0; k < 6; ++k) win[k] = win[k + 1];
        }
    }
}

// ---------- GEMM2: pre2[m][e] = bn2(sum_d s1[m][d]*lin_w[e][d]) ----------
// one wave -> 32x64 tile; waves = 8000 -> 1000 blocks
__global__ void gemm2_kernel(const _Float16* __restrict__ s1, const _Float16* __restrict__ linh,
                             const float* __restrict__ g, const float* __restrict__ bb,
                             const float* __restrict__ mm, const float* __restrict__ vv,
                             float* __restrict__ pre2) {
    const int lane = threadIdx.x & 31;
    const int wave = blockIdx.x * (blockDim.x >> 5) + (threadIdx.x >> 5);
    const int ng = wave & 3;
    const int mt = wave >> 2;
    const int m0 = mt * 32;
    const int n0 = ng * 64;
    const int half = lane >> 4;
    const int l15 = lane & 15;

    v8f acc0[4], acc1[4];
#pragma unroll
    for (int i = 0; i < 4; ++i) { acc0[i] = zero8(); acc1[i] = zero8(); }

    const _Float16* a0ptr = s1 + (size_t)(m0 + l15) * D_ + half * 8;
    const _Float16* a1ptr = a0ptr + (size_t)16 * D_;
#pragma unroll
    for (int kc = 0; kc < D_; kc += 32) {
        v16h a0 = load_frag(a0ptr + kc);
        v16h a1 = load_frag(a1ptr + kc);
#pragma unroll
        for (int nt = 0; nt < 4; ++nt) {
            const _Float16* bptr = linh + (size_t)(n0 + nt * 16 + l15) * D_ + half * 8 + kc;
            v16h bf = load_frag(bptr);
            acc0[nt] = __builtin_amdgcn_wmma_f32_16x16x32_f16(
                false, a0, false, bf, (short)0, acc0[nt], false, false);
            acc1[nt] = __builtin_amdgcn_wmma_f32_16x16x32_f16(
                false, a1, false, bf, (short)0, acc1[nt], false, false);
        }
    }
#pragma unroll
    for (int nt = 0; nt < 4; ++nt) {
        int col = n0 + nt * 16 + l15;
        float scale = g[col] * rsqrtf(vv[col] + EPS_);
        float bias  = bb[col] - mm[col] * scale;
#pragma unroll
        for (int r = 0; r < 8; ++r) {
            int row0 = m0 + r + 8 * half;
            pre2[(size_t)row0 * D_ + col]        = acc0[nt][r] * scale + bias;
            pre2[(size_t)(row0 + 16) * D_ + col] = acc1[nt][r] * scale + bias;
        }
    }
}

// ---------- LIF2 scan + residual (groups of 4) ----------
__global__ void lif2_res_kernel(const float* __restrict__ pre2, const _Float16* __restrict__ s1,
                                float* __restrict__ out) {
    int tid = blockIdx.x * blockDim.x + threadIdx.x;
    if (tid >= NBD) return;
    float v = 0.0f;
    for (int t0 = 0; t0 < T_; t0 += 4) {
        float xn[4], rn[4];
#pragma unroll
        for (int j = 0; j < 4; ++j) {
            size_t idx = (size_t)(t0 + j) * NBD + tid;
            xn[j] = pre2[idx];
            rn[j] = (float)s1[idx];
        }
        if (t0 + 8 < T_)
            __builtin_prefetch(pre2 + (size_t)(t0 + 8) * NBD + tid, 0, 1);
#pragma unroll
        for (int j = 0; j < 4; ++j) {
            v = v + (xn[j] - v) * TAU_INV;
            bool sp = (v - V_TH) >= 0.0f;
            v = sp ? 0.0f : v;
            out[(size_t)(t0 + j) * NBD + tid] = (sp ? 1.0f : 0.0f) + rn[j];
        }
    }
}

extern "C" void kernel_launch(void* const* d_in, const int* in_sizes, int n_in,
                              void* d_out, int out_size, void* d_ws, size_t ws_size,
                              hipStream_t stream) {
    const float* x     = (const float*)d_in[0];
    const float* pw_w  = (const float*)d_in[1];
    const float* pw_b  = (const float*)d_in[2];
    const float* dw_w  = (const float*)d_in[3];
    const float* bn1_g = (const float*)d_in[4];
    const float* bn1_b = (const float*)d_in[5];
    const float* bn1_m = (const float*)d_in[6];
    const float* bn1_v = (const float*)d_in[7];
    const float* lin_w = (const float*)d_in[8];
    const float* bn2_g = (const float*)d_in[9];
    const float* bn2_b = (const float*)d_in[10];
    const float* bn2_m = (const float*)d_in[11];
    const float* bn2_v = (const float*)d_in[12];
    float* out = (float*)d_out;

    // workspace layout (256B aligned)
    char* ws = (char*)d_ws;
    size_t off = 0;
    _Float16* xh   = (_Float16*)(ws + off); off += (size_t)M_ * KPAD1 * 2;   // 20.48 MB
    _Float16* pwh  = (_Float16*)(ws + off); off += (size_t)D_ * KPAD1 * 2;   // 80 KB
    _Float16* linh = (_Float16*)(ws + off); off += (size_t)D_ * D_ * 2;      // 128 KB
    _Float16* h    = (_Float16*)(ws + off); // (T,B,D) f16, 32.8 MB
    float*    pre2 = (float*)(ws + off);    // reuse region: h dead after dw_lif1
    off += (size_t)M_ * D_ * 4;             // region sized for the f32 pre2
    _Float16* s1   = (_Float16*)(ws + off); off += (size_t)M_ * D_ * 2;      // 32.8 MB

    // 1. conversions
    {
        int n = M_ * KPAD1;
        convert_x_kernel<<<(n + 255) / 256, 256, 0, stream>>>(x, xh);
        int nw = D_ * D_;  // covers both pw (40960) and lin (65536)
        convert_w_kernel<<<(nw + 255) / 256, 256, 0, stream>>>(pw_w, lin_w, pwh, linh);
    }
    // 2. pointwise GEMM (WMMA, 32x64/wave): 8000 waves
    gemm1_kernel<<<1000, 256, 0, stream>>>(xh, pwh, pw_b, h);
    // 3. depthwise conv + BN1 + LIF1 scan
    dw_lif1_kernel<<<NBD / 256, 256, 0, stream>>>(h, dw_w, bn1_g, bn1_b, bn1_m, bn1_v, s1);
    // 4. linear GEMM (WMMA, 32x64/wave) + BN2
    gemm2_kernel<<<1000, 256, 0, stream>>>(s1, linh, bn2_g, bn2_b, bn2_m, bn2_v, pre2);
    // 5. LIF2 scan + residual
    lif2_res_kernel<<<NBD / 256, 256, 0, stream>>>(pre2, s1, out);
}
